// DSTAGNN_block_59493886984767
// MI455X (gfx1250) — matrix operand order repaired and checked by reading, not compiled, not run on hardware
//
#include <hip/hip_runtime.h>
#include <math.h>

#define B_  16
#define N_  1024
#define T_  12
#define H_  3
#define DK_ 32
#define DM_ 512
#define K_  3
#define C_  64

typedef __bf16 bf16;
typedef __attribute__((ext_vector_type(16))) __bf16 v16bf;
typedef __attribute__((ext_vector_type(8)))  __bf16 v8bf;
typedef __attribute__((ext_vector_type(8)))  float  v8f;

// ---------------------------------------------------------------------------
// Stage 1: TEmx[b,t,n] = LN_n( x[b,n,0,t] + pos_T[t,n] ) * gT + bT
// grid = B*T blocks, 256 threads
// ---------------------------------------------------------------------------
__global__ void k_embedT(const float* __restrict__ x, const float* __restrict__ posT,
                         const float* __restrict__ gT, const float* __restrict__ bT,
                         float* __restrict__ TEmx) {
  int b = blockIdx.x / T_, t = blockIdx.x % T_;
  int tid = threadIdx.x;
  __shared__ float red[256];
  float v[4];
  float s = 0.f;
#pragma unroll
  for (int i = 0; i < 4; ++i) {
    int n = tid + 256 * i;
    v[i] = x[(b * N_ + n) * T_ + t] + posT[t * N_ + n];
    s += v[i];
  }
  red[tid] = s; __syncthreads();
  for (int off = 128; off > 0; off >>= 1) {
    if (tid < off) red[tid] += red[tid + off];
    __syncthreads();
  }
  float mu = red[0] * (1.f / N_);
  __syncthreads();
  s = 0.f;
#pragma unroll
  for (int i = 0; i < 4; ++i) { float d = v[i] - mu; s += d * d; }
  red[tid] = s; __syncthreads();
  for (int off = 128; off > 0; off >>= 1) {
    if (tid < off) red[tid] += red[tid + off];
    __syncthreads();
  }
  float rs = rsqrtf(red[0] * (1.f / N_) + 1e-5f);
#pragma unroll
  for (int i = 0; i < 4; ++i) {
    int n = tid + 256 * i;
    TEmx[(b * T_ + t) * N_ + n] = (v[i] - mu) * rs * gT[n] + bT[n];
  }
}

// ---------------------------------------------------------------------------
// Stage 2: temporal attention per batch. grid = B blocks, 256 threads.
// Writes scores to d_out tail and TATout (B,T,N) to workspace.
// dynamic LDS: 30048 floats
// ---------------------------------------------------------------------------
__global__ void k_tatt(const float* __restrict__ TEmx,
                       const float* __restrict__ wq, const float* __restrict__ wk,
                       const float* __restrict__ wv, const float* __restrict__ fc,
                       const float* __restrict__ res_att,
                       float* __restrict__ scores_out, float* __restrict__ TAT) {
  extern __shared__ float sm2[];
  float* te   = sm2;          // 12288
  float* outl = te + 12288;   // 12288
  float* qm   = outl + 12288; // 1152
  float* km   = qm + 1152;    // 1152
  float* vm   = km + 1152;    // 1152
  float* scl  = vm + 1152;    // 432
  float* atl  = scl + 432;    // 432
  float* cf   = atl + 432;    // 1152
  int b = blockIdx.x;
  int tid = threadIdx.x;
  const float inv = 0.17677669529663687f; // 1/sqrt(32)

  for (int i = tid; i < T_ * N_; i += 256) te[i] = TEmx[b * T_ * N_ + i];
  __syncthreads();

  // Q,K,V projections: (12,1024)@(1024,96)
  for (int idx = tid; idx < 3 * T_ * 96; idx += 256) {
    int which = idx / (T_ * 96), r = idx % (T_ * 96);
    int tq = r / 96, c = r % 96;
    const float* wgt = (which == 0) ? wq : (which == 1) ? wk : wv;
    const float* tr = &te[tq * N_];
    float s = 0.f;
    for (int n = 0; n < N_; ++n) s += tr[n] * wgt[n * 96 + c];
    float* dst = (which == 0) ? qm : (which == 1) ? km : vm;
    dst[r] = s;
  }
  __syncthreads();

  // scores (H,T,T) = Q.K^T/sqrt(32) + res_att  -> also written to output
  for (int idx = tid; idx < H_ * T_ * T_; idx += 256) {
    int h = idx / (T_ * T_), r = idx % (T_ * T_);
    int q = r / T_, kc = r % T_;
    float s = 0.f;
#pragma unroll
    for (int d2 = 0; d2 < DK_; ++d2)
      s += qm[q * 96 + h * DK_ + d2] * km[kc * 96 + h * DK_ + d2];
    s = s * inv + res_att[b * (H_ * T_ * T_) + idx];
    scl[idx] = s;
    scores_out[b * (H_ * T_ * T_) + idx] = s;
  }
  __syncthreads();

  // softmax over axis=3 (the query axis) as in the reference
  for (int idx = tid; idx < H_ * T_; idx += 256) {
    int h = idx / T_, kc = idx % T_;
    float mx = -1e30f;
    for (int q = 0; q < T_; ++q) mx = fmaxf(mx, scl[h * 144 + q * 12 + kc]);
    float Z = 0.f;
    for (int q = 0; q < T_; ++q) Z += __expf(scl[h * 144 + q * 12 + kc] - mx);
    float iZ = 1.f / Z;
    for (int q = 0; q < T_; ++q)
      atl[h * 144 + q * 12 + kc] = __expf(scl[h * 144 + q * 12 + kc] - mx) * iZ;
  }
  __syncthreads();

  // ctx (T, 96)
  for (int idx = tid; idx < T_ * 96; idx += 256) {
    int tq = idx / 96, c = idx % 96, h = c / DK_;
    float s = 0.f;
#pragma unroll
    for (int kc = 0; kc < T_; ++kc) s += atl[h * 144 + tq * 12 + kc] * vm[kc * 96 + c];
    cf[idx] = s;
  }
  __syncthreads();

  // ctx@fc + residual
  for (int idx = tid; idx < T_ * N_; idx += 256) {
    int tq = idx / N_, n = idx % N_;
    float s = te[idx];
#pragma unroll
    for (int c = 0; c < 96; ++c) s += cf[tq * 96 + c] * fc[c * N_ + n];
    outl[idx] = s;
  }
  __syncthreads();

  // LayerNorm over N (no affine), one wave per row
  int lane = tid & 31, w = tid >> 5;
  for (int row = w; row < T_; row += 8) {
    float s = 0.f;
    for (int n = lane; n < N_; n += 32) s += outl[row * N_ + n];
#pragma unroll
    for (int off = 16; off > 0; off >>= 1) s += __shfl_xor(s, off, 32);
    float mu = s * (1.f / N_);
    float v2 = 0.f;
    for (int n = lane; n < N_; n += 32) { float d = outl[row * N_ + n] - mu; v2 += d * d; }
#pragma unroll
    for (int off = 16; off > 0; off >>= 1) v2 += __shfl_xor(v2, off, 32);
    float rs = rsqrtf(v2 * (1.f / N_) + 1e-5f);
    for (int n = lane; n < N_; n += 32)
      TAT[(b * T_ + row) * N_ + n] = (outl[row * N_ + n] - mu) * rs;
  }
}

// ---------------------------------------------------------------------------
// Stage 3: pre_conv (T->DM per (b,n)) + pos_S + LN(DM) -> SEmx (B,N,DM)
// grid = B*N blocks, 256 threads
// ---------------------------------------------------------------------------
__global__ void k_preconv(const float* __restrict__ TAT, const float* __restrict__ pre_w,
                          const float* __restrict__ pre_b, const float* __restrict__ posS,
                          const float* __restrict__ gS, const float* __restrict__ bS,
                          float* __restrict__ SEmx) {
  int b = blockIdx.x >> 10, n = blockIdx.x & 1023;
  int tid = threadIdx.x;
  __shared__ float tat[T_];
  __shared__ float red[256];
  if (tid < T_) tat[tid] = TAT[(b * T_ + tid) * N_ + n];
  __syncthreads();
  float vv[2];
  float s = 0.f;
#pragma unroll
  for (int i = 0; i < 2; ++i) {
    int d = tid + 256 * i;
    float a = pre_b[d];
#pragma unroll
    for (int t = 0; t < T_; ++t) a += tat[t] * pre_w[d * T_ + t];
    vv[i] = a + posS[(size_t)n * DM_ + d];
    s += vv[i];
  }
  red[tid] = s; __syncthreads();
  for (int off = 128; off > 0; off >>= 1) { if (tid < off) red[tid] += red[tid + off]; __syncthreads(); }
  float mu = red[0] * (1.f / DM_);
  __syncthreads();
  s = 0.f;
#pragma unroll
  for (int i = 0; i < 2; ++i) { float d = vv[i] - mu; s += d * d; }
  red[tid] = s; __syncthreads();
  for (int off = 128; off > 0; off >>= 1) { if (tid < off) red[tid] += red[tid + off]; __syncthreads(); }
  float rs = rsqrtf(red[0] * (1.f / DM_) + 1e-5f);
#pragma unroll
  for (int i = 0; i < 2; ++i) {
    int d = tid + 256 * i;
    SEmx[((size_t)(b * N_ + n)) * DM_ + d] = (vv[i] - mu) * rs * gS[d] + bS[d];
  }
}

// ---------------------------------------------------------------------------
// Stage 4: SQ/SK = SEmx @ swq/swk, stored bf16 (B,N,96) for WMMA
// grid = B*N blocks, 192 threads (t<96 -> SQ, else SK)
// ---------------------------------------------------------------------------
__global__ void k_proj(const float* __restrict__ SEmx, const float* __restrict__ swq,
                       const float* __restrict__ swk,
                       bf16* __restrict__ SQ, bf16* __restrict__ SK) {
  int b = blockIdx.x >> 10, n = blockIdx.x & 1023;
  int tid = threadIdx.x;
  __shared__ float se[DM_];
  for (int i = tid; i < DM_; i += 192) se[i] = SEmx[((size_t)(b * N_ + n)) * DM_ + i];
  __syncthreads();
  int c = tid;
  const float* wgt = swq;
  bf16* dst = SQ;
  if (c >= 96) { c -= 96; wgt = swk; dst = SK; }
  float s = 0.f;
  for (int d = 0; d < DM_; ++d) s += se[d] * wgt[d * 96 + c];
  dst[(size_t)(b * N_ + n) * 96 + c] = (bf16)s;
}

// ---------------------------------------------------------------------------
// Stage 4b: convert GTU conv weights to bf16 (K index = ci*kw+dt, co-major)
// ---------------------------------------------------------------------------
__global__ void k_wcvt(const float* __restrict__ g3w, const float* __restrict__ g5w,
                       const float* __restrict__ g7w,
                       bf16* __restrict__ w3, bf16* __restrict__ w5, bf16* __restrict__ w7) {
  int i = blockIdx.x * 256 + threadIdx.x;
  if (i < 128 * 192) w3[i] = (bf16)g3w[i];
  if (i < 128 * 320) w5[i] = (bf16)g5w[i];
  if (i < 128 * 448) w7[i] = (bf16)g7w[i];
}

// ---------------------------------------------------------------------------
// Stage 5: fused spatial attention + Chebyshev graph conv (flash-style).
// rhs[b,k,m,t] = (sum_n cheb[k,n,m]*e^{s(n,m)-mx(m)}*x[b,n,t]) / sum_n e^{s(n,m)-mx(m)}
// s(n,m) = SQ[n].SK[m]/sqrt(32) + adj[n,m]*cmask[k,n,m]   (softmax over n)
// grid = dim3(N/16, K, B), 256 threads = 8 waves; WMMA bf16 16x16x32.
// ---------------------------------------------------------------------------
__global__ void k_spatial(const bf16* __restrict__ SQ, const bf16* __restrict__ SK,
                          const float* __restrict__ adj, const float* __restrict__ cmask,
                          const float* __restrict__ cheb, const float* __restrict__ x,
                          float* __restrict__ rhs) {
  __shared__ float xl[N_ * T_];      // 48 KB: x[b] tile
  __shared__ float cmb[8][16][14];   // per-wave online-softmax partials
  int m0 = blockIdx.x * 16, k = blockIdx.y, b = blockIdx.z;
  int tid = threadIdx.x;
  for (int i = tid; i < N_ * T_; i += 256) xl[i] = x[(size_t)b * N_ * T_ + i];
  __syncthreads();

  int lane = tid & 31, w = tid >> 5;
  int lm = lane & 15, lg = lane >> 4;

  // B operand: SK rows m0..m0+15 (ISA B layout: lane=col, K=8*lanegroup+e, 16+8*lanegroup+e)
  v16bf bB;
  {
    const bf16* p = SK + (size_t)(b * N_ + m0 + lm) * 96 + k * DK_ + lg * 8;
    v8bf b0 = *(const v8bf*)(p);
    v8bf b1 = *(const v8bf*)(p + 16);
#pragma unroll
    for (int e = 0; e < 8; ++e) { bB[e] = b0[e]; bB[8 + e] = b1[e]; }
  }

  const float inv = 0.17677669529663687f;
  float mx = -1e30f, Z = 0.f;
  float acc[T_];
#pragma unroll
  for (int t = 0; t < T_; ++t) acc[t] = 0.f;

  const size_t kNN = (size_t)k * N_ * N_;
  int mcol = m0 + lm;

  for (int it = 0; it < 8; ++it) {
    int n0 = (w + 8 * it) * 16;
    // prefetch next tile's bias/cheb streams (lowers to global_prefetch_b8)
    if (it < 7) {
      int nn = (w + 8 * (it + 1)) * 16 + 8 * lg;
      __builtin_prefetch(&adj[(size_t)nn * N_ + mcol], 0, 1);
      __builtin_prefetch(&cmask[kNN + (size_t)nn * N_ + mcol], 0, 1);
      __builtin_prefetch(&cheb[kNN + (size_t)nn * N_ + mcol], 0, 1);
    }
    // A operand: SQ rows n0..n0+15 (ISA 16-bit A layout)
    v16bf aA;
    {
      const bf16* p = SQ + (size_t)(b * N_ + n0 + lm) * 96 + k * DK_ + lg * 8;
      v8bf a0 = *(const v8bf*)(p);
      v8bf a1 = *(const v8bf*)(p + 16);
#pragma unroll
      for (int e = 0; e < 8; ++e) { aA[e] = a0[e]; aA[8 + e] = a1[e]; }
    }
    v8f c0 = {0.f, 0.f, 0.f, 0.f, 0.f, 0.f, 0.f, 0.f};
    v8f dte = __builtin_amdgcn_wmma_f32_16x16x32_bf16(false, aA, false, bB,
                                                      (short)0, c0, false, false);
    // lane holds column mcol, rows n0 + v + 8*lg
    float sv[8];
#pragma unroll
    for (int v = 0; v < 8; ++v) {
      int n = n0 + v + 8 * lg;
      size_t nm = (size_t)n * N_ + mcol;
      sv[v] = dte[v] * inv + adj[nm] * cmask[kNN + nm];
    }
    float tmax = sv[0];
#pragma unroll
    for (int v = 1; v < 8; ++v) tmax = fmaxf(tmax, sv[v]);
    float mnew = fmaxf(mx, tmax);
    float scale = __expf(mx - mnew);
    Z *= scale;
#pragma unroll
    for (int t = 0; t < T_; ++t) acc[t] *= scale;
#pragma unroll
    for (int v = 0; v < 8; ++v) {
      int n = n0 + v + 8 * lg;
      float e = __expf(sv[v] - mnew);
      Z += e;
      float wv = e * cheb[kNN + (size_t)n * N_ + mcol];
      const float* xr = &xl[n * T_];
#pragma unroll
      for (int t = 0; t < T_; ++t) acc[t] += wv * xr[t];
    }
    mx = mnew;
  }

  // combine lane L with L^16 (same column, other row-half)
  {
    float mo = __shfl_xor(mx, 16, 32);
    float Zo = __shfl_xor(Z, 16, 32);
    float mnew = fmaxf(mx, mo);
    float s1 = __expf(mx - mnew), s2 = __expf(mo - mnew);
    Z = Z * s1 + Zo * s2;
#pragma unroll
    for (int t = 0; t < T_; ++t) {
      float ao = __shfl_xor(acc[t], 16, 32);
      acc[t] = acc[t] * s1 + ao * s2;
    }
    mx = mnew;
  }
  if (lane < 16) {
    cmb[w][lane][0] = mx;
    cmb[w][lane][1] = Z;
#pragma unroll
    for (int t = 0; t < T_; ++t) cmb[w][lane][2 + t] = acc[t];
  }
  __syncthreads();

  if (tid < 16) {
    int m = tid;
    float M = cmb[0][m][0], ZZ = cmb[0][m][1], a[T_];
#pragma unroll
    for (int t = 0; t < T_; ++t) a[t] = cmb[0][m][2 + t];
    for (int ww = 1; ww < 8; ++ww) {
      float mo = cmb[ww][m][0];
      float mn = fmaxf(M, mo);
      float s1 = __expf(M - mn), s2 = __expf(mo - mn);
      ZZ = ZZ * s1 + cmb[ww][m][1] * s2;
#pragma unroll
      for (int t = 0; t < T_; ++t) a[t] = a[t] * s1 + cmb[ww][m][2 + t] * s2;
      M = mn;
    }
    float iZ = 1.f / ZZ;
    float* rp = rhs + ((size_t)(b * K_ + k) * N_ + m0 + m) * T_;
#pragma unroll
    for (int t = 0; t < T_; ++t) rp[t] = a[t] * iZ;
  }
}

// ---------------------------------------------------------------------------
// Stage 6: theta contraction + GTU convs (WMMA im2col) + fc + residual + LN.
// grid = dim3(N/4, B), 256 threads (8 waves), G=4 nodes per block.
// Conv as GEMM: y(128, pos) = W(128, ci*kw) @ P(ci*kw, pos), bf16 WMMA.
// ---------------------------------------------------------------------------
__global__ void k_gtu(const float* __restrict__ rhs, const float* __restrict__ theta,
                      const bf16* __restrict__ w3, const float* __restrict__ g3b,
                      const bf16* __restrict__ w5, const float* __restrict__ g5b,
                      const bf16* __restrict__ w7, const float* __restrict__ g7b,
                      const float* __restrict__ fcw, const float* __restrict__ fcb,
                      const float* __restrict__ rw, const float* __restrict__ rb,
                      const float* __restrict__ gln, const float* __restrict__ bln,
                      const float* __restrict__ x, float* __restrict__ out) {
  extern __shared__ float sm6[];
  float* rl  = sm6;            // 144   rhs local [k*48 + j*12 + t]
  float* Xl  = rl + 144;       // 3072  [c*48 + j*12 + t]
  float* yb  = Xl + 3072;      // 5120  [co*40 + pos]
  float* tcb = yb + 5120;      // 6144  [c*96 + j*24 + jj]
  float* pre = tcb + 6144;     // 3072  [c*48 + j*12 + t]
  float* mrs = pre + 3072;     // 96
  bf16*  Pb  = (bf16*)(mrs + 96); // im2col patches, up to 14592 bf16

  int n0 = blockIdx.x * 4, b = blockIdx.y;
  int tid = threadIdx.x;
  int lane = tid & 31, w = tid >> 5;
  int lm = lane & 15, lg = lane >> 4;

  for (int i = tid; i < 144; i += 256) {
    int kk = i / 48, r = i % 48;
    int j = r / 12, t = r % 12;
    rl[i] = rhs[((size_t)(b * K_ + kk) * N_ + n0 + j) * T_ + t];
  }
  __syncthreads();
  // spatial_gcn -> X[c][j][t] = relu(sum_k rhs[k][j][t]*theta[k][c])
  for (int i = tid; i < 3072; i += 256) {
    int c = i / 48, r = i % 48;
    float s = rl[r] * theta[c] + rl[48 + r] * theta[64 + c] + rl[96 + r] * theta[128 + c];
    Xl[i] = fmaxf(s, 0.f);
  }
  __syncthreads();

  const bf16* Ws[3] = {w3, w5, w7};
  const float* Bs[3] = {g3b, g5b, g7b};
  const int kws[3]   = {3, 5, 7};
  const int tcs[3]   = {10, 8, 6};
  const int toff[3]  = {0, 10, 18};
  const int Ks[3]    = {192, 320, 448};
  const int sps[3]   = {200, 328, 456};   // padded P row stride (16B aligned)
  const int nposR[3] = {40, 32, 24};
  const int ntls[3]  = {3, 2, 2};         // N-tiles of 16 positions (padded)

  for (int cv = 0; cv < 3; ++cv) {
    int kw = kws[cv], tcn = tcs[cv];
    int Kd = Ks[cv], sp = sps[cv], npos = nposR[cv], ntl = ntls[cv];

    // build im2col patch matrix: P[pos][q], q = ci*kw+dt, pos = j*tcn+tau
    for (int i = tid; i < ntl * 16 * sp; i += 256) Pb[i] = (bf16)0.f;
    __syncthreads();
    for (int i = tid; i < npos * Kd; i += 256) {
      int pos = i / Kd, q = i % Kd;
      int j = pos / tcn, ta = pos % tcn;
      int ci = q / kw, dt = q % kw;
      Pb[pos * sp + q] = (bf16)Xl[ci * 48 + j * 12 + ta + dt];
    }
    __syncthreads();

    // WMMA tasks: 8 M-tiles x ntl N-tiles, uniform count per wave
    int ntask = 8 * ntl;
    for (int task = w; task < ntask; task += 8) {
      int mt = task / ntl, nt = task % ntl;
      v8f accv = {0.f, 0.f, 0.f, 0.f, 0.f, 0.f, 0.f, 0.f};
      for (int kc = 0; kc < Kd / 32; ++kc) {
        v16bf aA, bBv;
        {
          const bf16* ap = Ws[cv] + (size_t)(mt * 16 + lm) * Kd + kc * 32 + lg * 8;
          v8bf a0 = *(const v8bf*)(ap);
          v8bf a1 = *(const v8bf*)(ap + 16);
#pragma unroll
          for (int e = 0; e < 8; ++e) { aA[e] = a0[e]; aA[8 + e] = a1[e]; }
        }
        {
          const bf16* bp = &Pb[(nt * 16 + lm) * sp + kc * 32 + lg * 8];
          v8bf b0 = *(const v8bf*)(bp);
          v8bf b1 = *(const v8bf*)(bp + 16);
#pragma unroll
          for (int e = 0; e < 8; ++e) { bBv[e] = b0[e]; bBv[8 + e] = b1[e]; }
        }
        accv = __builtin_amdgcn_wmma_f32_16x16x32_bf16(false, aA, false, bBv,
                                                       (short)0, accv, false, false);
      }
      int pos = nt * 16 + lm;
      if (pos < npos) {
#pragma unroll
        for (int v = 0; v < 8; ++v) {
          int co2 = mt * 16 + v + 8 * lg;
          yb[co2 * 40 + pos] = accv[v] + Bs[cv][co2];
        }
      }
    }
    __syncthreads();

    // gate: tanh(y[:C]) * sigmoid(y[C:])
    for (int i = tid; i < C_ * 4 * tcn; i += 256) {
      int c = i / (4 * tcn), pos = i % (4 * tcn);
      int j = pos / tcn, ta = pos % tcn;
      float a = yb[c * 40 + pos];
      float g = yb[(c + C_) * 40 + pos];
      tcb[c * 96 + j * 24 + toff[cv] + ta] = tanhf(a) * (1.f / (1.f + __expf(-g)));
    }
    __syncthreads();
  }

  // time_conv (24->12) + relu + residual + relu
  for (int i = tid; i < 3072; i += 256) {
    int c = i / 48, r = i % 48, j = r / 12, t = r % 12;
    float s = fcb[t];
    const float* tr = &tcb[c * 96 + j * 24];
#pragma unroll
    for (int jj = 0; jj < 24; ++jj) s += tr[jj] * fcw[jj * 12 + t];
    s = fmaxf(s, 0.f);
    float resid = x[((size_t)b * N_ + n0 + j) * T_ + t] * rw[c] + rb[c];
    pre[c * 48 + r] = fmaxf(resid + s, 0.f);
  }
  __syncthreads();
  // LN over the 64 channels for each (j,t)
  if (tid < 48) {
    float s = 0.f;
    for (int c = 0; c < C_; ++c) s += pre[c * 48 + tid];
    float mu = s * (1.f / C_);
    float v2 = 0.f;
    for (int c = 0; c < C_; ++c) { float d = pre[c * 48 + tid] - mu; v2 += d * d; }
    mrs[tid * 2] = mu;
    mrs[tid * 2 + 1] = rsqrtf(v2 * (1.f / C_) + 1e-5f);
  }
  __syncthreads();
  size_t ob = ((size_t)b * N_ + n0) * (C_ * T_);
  for (int i = tid; i < 3072; i += 256) {
    int j = i / 768, r = i % 768, c = r / 12, t = r % 12;
    float mu = mrs[(j * 12 + t) * 2], rs = mrs[(j * 12 + t) * 2 + 1];
    out[ob + i] = (pre[c * 48 + j * 12 + t] - mu) * rs * gln[c] + bln[c];
  }
}

// ---------------------------------------------------------------------------
extern "C" void kernel_launch(void* const* d_in, const int* in_sizes, int n_in,
                              void* d_out, int out_size, void* d_ws, size_t ws_size,
                              hipStream_t stream) {
  const float* x       = (const float*)d_in[0];
  const float* res_att = (const float*)d_in[1];
  const float* pos_T   = (const float*)d_in[2];
  const float* gT      = (const float*)d_in[3];
  const float* bT      = (const float*)d_in[4];
  const float* wq      = (const float*)d_in[5];
  const float* wk      = (const float*)d_in[6];
  const float* wv      = (const float*)d_in[7];
  const float* fc      = (const float*)d_in[8];
  const float* pre_w   = (const float*)d_in[9];
  const float* pre_b   = (const float*)d_in[10];
  const float* pos_S   = (const float*)d_in[11];
  const float* gS      = (const float*)d_in[12];
  const float* bS      = (const float*)d_in[13];
  const float* swq     = (const float*)d_in[14];
  const float* swk     = (const float*)d_in[15];
  const float* cheb    = (const float*)d_in[16];
  const float* adj_pa  = (const float*)d_in[17];
  const float* cmask   = (const float*)d_in[18];
  const float* theta   = (const float*)d_in[19];
  const float* g3w     = (const float*)d_in[20];
  const float* g3b     = (const float*)d_in[21];
  const float* g5w     = (const float*)d_in[22];
  const float* g5b     = (const float*)d_in[23];
  const float* g7w     = (const float*)d_in[24];
  const float* g7b     = (const float*)d_in[25];
  const float* fcw     = (const float*)d_in[26];
  const float* fcb     = (const float*)d_in[27];
  const float* rw      = (const float*)d_in[28];
  const float* rb      = (const float*)d_in[29];
  const float* gln     = (const float*)d_in[30];
  const float* bln     = (const float*)d_in[31];

  float* out = (float*)d_out;
  const int OUT0 = B_ * N_ * C_ * T_; // 12582912; scores follow

  // workspace layout
  float* TEmx = (float*)d_ws;                         // 196608 f
  float* TAT  = TEmx + (size_t)B_ * T_ * N_;          // 196608 f
  float* SEmx = TAT + (size_t)B_ * T_ * N_;           // 8388608 f
  bf16*  SQ   = (bf16*)(SEmx + (size_t)B_ * N_ * DM_);// 1572864 bf16
  bf16*  SK   = SQ + (size_t)B_ * N_ * 96;            // 1572864 bf16
  float* rhs  = (float*)(SK + (size_t)B_ * N_ * 96);  // 589824 f
  bf16*  w3   = (bf16*)(rhs + (size_t)B_ * K_ * N_ * T_); // 24576 bf16
  bf16*  w5   = w3 + 128 * 192;                       // 40960 bf16
  bf16*  w7   = w5 + 128 * 320;                       // 57344 bf16

  k_embedT<<<B_ * T_, 256, 0, stream>>>(x, pos_T, gT, bT, TEmx);

  size_t sh2 = 30048 * sizeof(float);
  k_tatt<<<B_, 256, sh2, stream>>>(TEmx, wq, wk, wv, fc, res_att, out + OUT0, TAT);

  k_preconv<<<B_ * N_, 256, 0, stream>>>(TAT, pre_w, pre_b, pos_S, gS, bS, SEmx);

  k_proj<<<B_ * N_, 192, 0, stream>>>(SEmx, swq, swk, SQ, SK);

  k_wcvt<<<(128 * 448 + 255) / 256, 256, 0, stream>>>(g3w, g5w, g7w, w3, w5, w7);

  k_spatial<<<dim3(N_ / 16, K_, B_), 256, 0, stream>>>(SQ, SK, adj_pa, cmask, cheb, x, rhs);

  size_t sh6 = (17648 + 7296) * sizeof(float);
  k_gtu<<<dim3(N_ / 4, B_), 256, sh6, stream>>>(rhs, theta, w3, g3b, w5, g5b,
                                                w7, g7b, fcw, fcb, rw, rb,
                                                gln, bln, x, out);
}
